// InfoNCELoss_22316650070694
// MI455X (gfx1250) — matrix operand order repaired and checked by reading, not compiled, not run on hardware
//
#include <hip/hip_runtime.h>

typedef __attribute__((ext_vector_type(16))) _Float16 v16h;
typedef __attribute__((ext_vector_type(8)))  _Float16 v8h;
typedef __attribute__((ext_vector_type(4)))  _Float16 v4h;
typedef __attribute__((ext_vector_type(8)))  float    v8f;
typedef __attribute__((ext_vector_type(4)))  int      v4i;

#define N_ROWS 16384
#define DIM    128
#define INV_T  5.0f                     // 1 / 0.2
#define SCALE2 7.213475204444817f       // (1/T) * log2(e): exp(x/T) = exp2(x*SCALE2)
#define EPSN   1e-12f
#define SPLIT  4                        // column splits (grid.y)
#define CHUNK  128                      // v2 rows staged in LDS per iteration
#define LDS_STRIDE 136                  // 128 + 8 f16 pad -> 272B row stride
#define BUF_ELEMS (CHUNK * LDS_STRIDE)  // one LDS buffer (f16 elems)

#define HAS_ASYNC_LDS __has_builtin(__builtin_amdgcn_global_load_async_to_lds_b128)

typedef __attribute__((address_space(1))) v4i* gv4i_p;   // global int4*
typedef __attribute__((address_space(3))) v4i* lv4i_p;   // LDS int4*

__device__ __forceinline__ void wait_async_lds() {
#if __has_builtin(__builtin_amdgcn_s_wait_asynccnt)
  __builtin_amdgcn_s_wait_asynccnt(0);
#else
  asm volatile("s_wait_asynccnt 0x0" ::: "memory");
#endif
}

// Stage CHUNK rows (128 x 256B = 32KB) of v2h into an LDS buffer.
// 256 threads: 2 threads/row, 8 x 16B per thread.
__device__ __forceinline__ void stage_chunk(const _Float16* __restrict__ gsrc,
                                            _Float16* ldst, int tid) {
  int row  = tid >> 1;
  int colh = (tid & 1) * 64;
  const _Float16* src = gsrc + (size_t)row * DIM + colh;
  _Float16*       dst = ldst + row * LDS_STRIDE + colh;
#if HAS_ASYNC_LDS
  _Float16* srcnc = const_cast<_Float16*>(src);
  #pragma unroll
  for (int j = 0; j < 8; ++j)
    __builtin_amdgcn_global_load_async_to_lds_b128(
        (gv4i_p)(srcnc + j * 8), (lv4i_p)(dst + j * 8), 0, 0);
#else
  uint4 t[8];
  #pragma unroll
  for (int j = 0; j < 8; ++j) t[j] = *(const uint4*)(src + j * 8);
  #pragma unroll
  for (int j = 0; j < 8; ++j) *(uint4*)(dst + j * 8) = t[j];
#endif
}

// ---------------------------------------------------------------------------
// Kernel 1: L2-normalize both views (fp32), emit f16 rows + positive dot.
// One wave (32 lanes) per row; 4 floats per lane.
// ---------------------------------------------------------------------------
__global__ __launch_bounds__(256) void infonce_normalize(
    const float* __restrict__ v1, const float* __restrict__ v2,
    _Float16* __restrict__ v1h, _Float16* __restrict__ v2h,
    float* __restrict__ pos_dot) {
  int gwave = (blockIdx.x * blockDim.x + threadIdx.x) >> 5;   // = row
  int lane  = threadIdx.x & 31;
  if (gwave >= N_ROWS) return;

  float4 a = ((const float4*)(v1 + (size_t)gwave * DIM))[lane];
  float4 b = ((const float4*)(v2 + (size_t)gwave * DIM))[lane];

  float s1 = a.x*a.x + a.y*a.y + a.z*a.z + a.w*a.w;
  float s2 = b.x*b.x + b.y*b.y + b.z*b.z + b.w*b.w;
  #pragma unroll
  for (int off = 16; off > 0; off >>= 1) {
    s1 += __shfl_xor(s1, off, 32);
    s2 += __shfl_xor(s2, off, 32);
  }
  float i1 = 1.0f / fmaxf(sqrtf(s1), EPSN);
  float i2 = 1.0f / fmaxf(sqrtf(s2), EPSN);
  a.x *= i1; a.y *= i1; a.z *= i1; a.w *= i1;
  b.x *= i2; b.y *= i2; b.z *= i2; b.w *= i2;

  float d = a.x*b.x + a.y*b.y + a.z*b.z + a.w*b.w;
  #pragma unroll
  for (int off = 16; off > 0; off >>= 1) d += __shfl_xor(d, off, 32);

  v4h ha = { (_Float16)a.x, (_Float16)a.y, (_Float16)a.z, (_Float16)a.w };
  v4h hb = { (_Float16)b.x, (_Float16)b.y, (_Float16)b.z, (_Float16)b.w };
  ((v4h*)(v1h + (size_t)gwave * DIM))[lane] = ha;
  ((v4h*)(v2h + (size_t)gwave * DIM))[lane] = hb;
  if (lane == 0) pos_dot[gwave] = d;
}

// ---------------------------------------------------------------------------
// Kernel 2: partial row sums of exp(sim/T) via WMMA f16->f32.
// grid = (N/128, SPLIT). Block = 8 waves; each wave owns a 16-row strip
// (A frags resident in VGPRs) and sweeps this block's 4096-column slice.
// Double-buffered LDS chunks filled with async global->LDS DMA; B fragments
// software-pipelined in registers.
// ---------------------------------------------------------------------------
__global__ __launch_bounds__(256) void infonce_simexp(
    const _Float16* __restrict__ v1h, const _Float16* __restrict__ v2h,
    float* __restrict__ ttl_part) {
  __shared__ _Float16 lds[2 * BUF_ELEMS];   // 2 x 34816 B

  const int tid  = threadIdx.x;
  const int wave = tid >> 5;
  const int lane = tid & 31;
  const int half = lane >> 4;        // K-half selector per WMMA layout
  const int lr   = lane & 15;        // row (A) / column (B) within tile
  const int m_base = blockIdx.x * 128 + wave * 16;
  const int col0   = blockIdx.y * (N_ROWS / SPLIT);
  const int nchunks = (N_ROWS / SPLIT) / CHUNK;   // 32

  // --- A fragments: 16x32 f16 layout (resident for whole kernel) ---
  // lanes 0-15 : row M=lr, elems 0-7 = K k0+0..7,  elems 8-15 = K k0+16..23
  // lanes 16-31: row M=lr, elems 0-7 = K k0+8..15, elems 8-15 = K k0+24..31
  v16h afrag[4];
  {
    const _Float16* arow = v1h + (size_t)(m_base + lr) * DIM;
    #pragma unroll
    for (int kb = 0; kb < 4; ++kb) {
      int k0 = kb * 32 + half * 8;
      union { v16h v; v8h h[2]; } u;
      u.h[0] = *(const v8h*)(arow + k0);
      u.h[1] = *(const v8h*)(arow + k0 + 16);
      afrag[kb] = u.v;
    }
  }

  float acc[8];
  #pragma unroll
  for (int r = 0; r < 8; ++r) acc[r] = 0.0f;

  // prologue: stage chunk 0 into buffer 0
  stage_chunk(v2h + (size_t)col0 * DIM, lds, tid);
  wait_async_lds();
  __syncthreads();

  for (int c = 0; c < nchunks; ++c) {
    // prefetch next chunk into the other buffer (overlaps with WMMA below)
    if (c + 1 < nchunks)
      stage_chunk(v2h + (size_t)(col0 + (c + 1) * CHUNK) * DIM,
                  lds + ((c + 1) & 1) * BUF_ELEMS, tid);

    // --- B fragments: 32x16 f16 layout ---
    // lane = column N=lr; lanes 0-15 hold K k0..k0+7, k0+16..k0+23; lanes 16-31
    // hold K k0+8..k0+15, k0+24..k0+31 (matches A layout).
    const _Float16* bbase = lds + (c & 1) * BUF_ELEMS + lr * LDS_STRIDE + half * 8;

    v16h b0[4], b1[4];
    #pragma unroll
    for (int kb = 0; kb < 4; ++kb) {
      union { v16h v; v8h h[2]; } b;
      b.h[0] = *(const v8h*)(bbase + kb * 32);
      b.h[1] = *(const v8h*)(bbase + kb * 32 + 16);
      b0[kb] = b.v;
    }

    #pragma unroll
    for (int ct = 0; ct < CHUNK / 16; ++ct) {
      if (ct + 1 < CHUNK / 16) {        // prefetch next tile's B fragments
        const _Float16* bn = bbase + (ct + 1) * 16 * LDS_STRIDE;
        #pragma unroll
        for (int kb = 0; kb < 4; ++kb) {
          union { v16h v; v8h h[2]; } b;
          b.h[0] = *(const v8h*)(bn + kb * 32);
          b.h[1] = *(const v8h*)(bn + kb * 32 + 16);
          b1[kb] = b.v;
        }
      }
      v8f cc = {};
      #pragma unroll
      for (int kb = 0; kb < 4; ++kb)
        cc = __builtin_amdgcn_wmma_f32_16x16x32_f16(
                 false, afrag[kb], false, b0[kb], (short)0, cc, false, false);
      // C layout: VGPR r, lanes 0-15 -> row r, lanes 16-31 -> row r+8, N=lr
      #pragma unroll
      for (int r = 0; r < 8; ++r)
        acc[r] += __builtin_amdgcn_exp2f(cc[r] * SCALE2);
      if (ct + 1 < CHUNK / 16) {
        #pragma unroll
        for (int kb = 0; kb < 4; ++kb) b0[kb] = b1[kb];
      }
    }

    wait_async_lds();      // my prefetch writes for chunk c+1 are done
    __syncthreads();       // everyone done writing next buf / reading cur buf
  }

  // reduce each row-sum across the 16 lanes of its half-group
  #pragma unroll
  for (int r = 0; r < 8; ++r) {
    float v = acc[r];
    v += __shfl_xor(v, 1, 32);
    v += __shfl_xor(v, 2, 32);
    v += __shfl_xor(v, 4, 32);
    v += __shfl_xor(v, 8, 32);
    if (lr == 0)           // rows exclusively owned within this column split
      ttl_part[(size_t)blockIdx.y * N_ROWS + m_base + half * 8 + r] = v;
  }
}

// ---------------------------------------------------------------------------
// Kernel 3: loss = mean( log(sum_p ttl_part[p][i]) - dot_i / T )
// Deterministic: fixed summation order over the SPLIT partials.
// ---------------------------------------------------------------------------
__global__ __launch_bounds__(256) void infonce_loss(
    const float* __restrict__ pos_dot, const float* __restrict__ ttl_part,
    float* __restrict__ out) {
  __shared__ float red[256];
  float s = 0.0f;
  for (int i = threadIdx.x; i < N_ROWS; i += 256) {
    float t = 0.0f;
    #pragma unroll
    for (int p = 0; p < SPLIT; ++p) t += ttl_part[(size_t)p * N_ROWS + i];
    s += __logf(t) - pos_dot[i] * INV_T;
  }
  red[threadIdx.x] = s;
  __syncthreads();
  #pragma unroll
  for (int off = 128; off > 0; off >>= 1) {
    if (threadIdx.x < off) red[threadIdx.x] += red[threadIdx.x + off];
    __syncthreads();
  }
  if (threadIdx.x == 0) out[0] = red[0] / (float)N_ROWS;
}

// ---------------------------------------------------------------------------
extern "C" void kernel_launch(void* const* d_in, const int* in_sizes, int n_in,
                              void* d_out, int out_size, void* d_ws, size_t ws_size,
                              hipStream_t stream) {
  const float* v1 = (const float*)d_in[0];
  const float* v2 = (const float*)d_in[1];

  // workspace layout
  _Float16* v1h  = (_Float16*)d_ws;                              // 4 MB
  _Float16* v2h  = v1h + (size_t)N_ROWS * DIM;                   // 4 MB
  float* pos     = (float*)(v2h + (size_t)N_ROWS * DIM);         // 64 KB
  float* ttl_p   = pos + N_ROWS;                                 // SPLIT * 64 KB

  // 1 wave per row, 8 rows per 256-thread block
  infonce_normalize<<<N_ROWS / 8, 256, 0, stream>>>(v1, v2, v1h, v2h, pos);

  // 128 row-blocks x 4 column-splits = 512 workgroups
  dim3 grid(N_ROWS / 128, SPLIT);
  infonce_simexp<<<grid, 256, 0, stream>>>(v1h, v2h, ttl_p);

  infonce_loss<<<1, 256, 0, stream>>>(pos, ttl_p, (float*)d_out);
}
